// Grid_SE3_18580028522892
// MI455X (gfx1250) — matrix operand order repaired and checked by reading, not compiled, not run on hardware
//
#include <hip/hip_runtime.h>
#include <math.h>

// ---------------------------------------------------------------------------
// SE(3)-equivariant GNN layer pipeline for MI455X (gfx1250, wave32, WMMA).
// All dense GEMMs use v_wmma_f32_16x16x32_bf16 (bf16 operands, f32 accum).
// Native bf16 converts (v_cvt_pk_bf16_f32) + b128 global loads for A tiles.
// Workspace requirement: ~670 MB (m0/m1/m2 per-edge buffers dominate).
// ---------------------------------------------------------------------------

#define NN 30000
#define EE 480000
#define CC 32
#define LL 2
#define PP 7
#define HH 4
#define EPSF 1e-8f

typedef __attribute__((ext_vector_type(16))) __bf16 v16bf;
typedef __attribute__((ext_vector_type(8)))  float  v8f;

// Native conversion: clang lowers to v_cvt (pairs pack into v_cvt_pk_bf16_f32).
__device__ __forceinline__ __bf16 f2bf(float f) { return (__bf16)f; }

__device__ __forceinline__ v8f wmma_bf16(v16bf a, v16bf b, v8f c) {
    // D = A(16x32) * B(32x16) + C ; f32 accumulate
    return __builtin_amdgcn_wmma_f32_16x16x32_bf16(false, a, false, b, (short)0, c, false, false);
}

// B tile builder: B[k][n] = W[k*stride + n0 + n], matching A layout:
// lanes 0-15 hold K 0-7 & 16-23 of column n=lane, lanes 16-31 hold K 8-15 & 24-31.
__device__ __forceinline__ v16bf loadB(const float* __restrict__ W, int stride, int n0, int lane) {
    v16bf b;
    const int n  = n0 + (lane & 15);
    const int kb = (lane >> 4) * 8;
    #pragma unroll
    for (int j = 0; j < 16; ++j) b[j] = f2bf(W[(size_t)(kb + j + (j & 8)) * stride + n]);
    return b;
}

// A tile builder from 16 contiguous row-major f32 rows of width 32.
// ISA 16-bit A layout: lanes 0-15: K 0-7 & 16-23 of row M=lane;
// lanes 16-31: K 8-15 & 24-31 of row M=lane-16.
// Row base is 128B aligned; kh*4 keeps 16B alignment -> b128 loads.
__device__ __forceinline__ v16bf loadA_row(const float* __restrict__ row, int lane) {
    v16bf a;
    const int kh = (lane >> 4) * 8;
    const float4* p0 = (const float4*)(row + kh);
    const float4* p1 = (const float4*)(row + kh + 16);
    const float4 c0 = p0[0], c1 = p0[1], c2 = p1[0], c3 = p1[1];
    a[0]  = f2bf(c0.x); a[1]  = f2bf(c0.y); a[2]  = f2bf(c0.z); a[3]  = f2bf(c0.w);
    a[4]  = f2bf(c1.x); a[5]  = f2bf(c1.y); a[6]  = f2bf(c1.z); a[7]  = f2bf(c1.w);
    a[8]  = f2bf(c2.x); a[9]  = f2bf(c2.y); a[10] = f2bf(c2.z); a[11] = f2bf(c2.w);
    a[12] = f2bf(c3.x); a[13] = f2bf(c3.y); a[14] = f2bf(c3.z); a[15] = f2bf(c3.w);
    return a;
}
__device__ __forceinline__ v16bf loadA_rows(const float* __restrict__ A, int lane) {
    return loadA_row(A + (size_t)(lane & 15) * CC, lane);
}

// D layout store: VGPR i, lane<16 -> row i, lane>=16 -> row i+8; col = lane&15.
__device__ __forceinline__ void storeD(float* __restrict__ out, int t, int lane, v8f d) {
    const int col = t * 16 + (lane & 15);
    const int hi  = lane >> 4;
    #pragma unroll
    for (int i = 0; i < 8; ++i) out[(size_t)(i + hi * 8) * CC + col] = d[i];
}

__device__ __forceinline__ void atomicMaxF(float* addr, float val) {
    if (val >= 0.0f) atomicMax((int*)addr, __float_as_int(val));
    else             atomicMin((unsigned int*)addr, __float_as_uint(val));
}

// ---------------------------------------------------------------------------
// Pass 1 (16 edges per wave): radial MLP (WMMA), messages, q/k (WMMA),
// logits + segment-max.
// f1 stored (N,3,C); f2 stored (N,5,C). m1 stored (E,3,C); m2 (E,5,C).
// ---------------------------------------------------------------------------
#define WV 2   // waves per block
__global__ __launch_bounds__(WV * 32) void edge_pass1(
    const float* __restrict__ pos, const float* __restrict__ f0,
    const float* __restrict__ f1,  const float* __restrict__ f2,
    const float* __restrict__ edge_feat,
    const int* __restrict__ esrc,  const int* __restrict__ edst,
    const float* __restrict__ Wr1, const float* __restrict__ br1,
    const float* __restrict__ Wr2, const float* __restrict__ Wq,
    const float* __restrict__ Wk,
    float* __restrict__ m0g, float* __restrict__ m1g, float* __restrict__ m2g,
    float* __restrict__ logitsg, float* __restrict__ mx)
{
    __shared__ float  geo [WV][16][9];
    __shared__ __bf16 hid [WV][16][32];
    __shared__ float  wbuf[WV][16][224];
    __shared__ float  m0s [WV][16][32];
    __shared__ float  qs  [WV][16][32];
    __shared__ float  ks  [WV][16][32];

    const int wid   = threadIdx.x >> 5;
    const int lane  = threadIdx.x & 31;
    const int ebase = (blockIdx.x * WV + wid) * 16;
    const int mrow  = lane & 15;
    const int hi    = lane >> 4;
    const int kh    = hi * 8;

    // ---- geometry (lanes 0..15, one edge each) ----
    if (lane < 16) {
        const int eg = ebase + lane;
        const int s = esrc[eg], d = edst[eg];
        const float rx = pos[(size_t)d*3+0] - pos[(size_t)s*3+0];
        const float ry = pos[(size_t)d*3+1] - pos[(size_t)s*3+1];
        const float rz = pos[(size_t)d*3+2] - pos[(size_t)s*3+2];
        const float r  = sqrtf(rx*rx + ry*ry + rz*rz + EPSF);
        const float iv = 1.0f / r;
        const float x = rx*iv, y = ry*iv, z = rz*iv;
        geo[wid][lane][0] = r;
        geo[wid][lane][1] = x;  geo[wid][lane][2] = y;  geo[wid][lane][3] = z;
        geo[wid][lane][4] = x*y; geo[wid][lane][5] = y*z;
        geo[wid][lane][6] = 3.0f*z*z - 1.0f;
        geo[wid][lane][7] = x*z; geo[wid][lane][8] = x*x - y*y;
    }
    __syncthreads();

    // ---- GEMM1: hidden = relu(edge_feat @ Wr1[1:] + r*Wr1[0] + br1) ----
    v16bf aF = loadA_row(edge_feat + (size_t)(ebase + mrow) * CC, lane);
    {
        v8f z = {};
        v8f d0 = wmma_bf16(aF, loadB(Wr1 + CC, CC,  0, lane), z);
        v8f d1 = wmma_bf16(aF, loadB(Wr1 + CC, CC, 16, lane), z);
        const int nc = lane & 15;
        const float w00 = Wr1[nc],      w01 = Wr1[16 + nc];
        const float b0  = br1[nc],      b1  = br1[16 + nc];
        #pragma unroll
        for (int i = 0; i < 8; ++i) {
            const int m = i + hi * 8;
            const float r = geo[wid][m][0];
            hid[wid][m][nc]      = f2bf(fmaxf(d0[i] + b0 + r * w00, 0.0f));
            hid[wid][m][16 + nc] = f2bf(fmaxf(d1[i] + b1 + r * w01, 0.0f));
        }
    }
    __syncthreads();

    // ---- GEMM2: w = hidden @ Wr2 (32 -> 224), 14 N-tiles ----
    v16bf aH;
    #pragma unroll
    for (int j = 0; j < 8; ++j) { aH[j] = hid[wid][mrow][kh + j]; aH[8+j] = hid[wid][mrow][kh + 16 + j]; }
    #pragma unroll
    for (int t = 0; t < 14; ++t) {
        v8f z = {};
        v8f d = wmma_bf16(aH, loadB(Wr2, PP * CC, t * 16, lane), z);
        const int nc = t * 16 + (lane & 15);
        #pragma unroll
        for (int i = 0; i < 8; ++i) wbuf[wid][i + hi * 8][nc] = d[i];
    }
    __syncthreads();

    // ---- messages: lane = channel, loop over 16 edges ----
    {
        const int c = lane;
        for (int e = 0; e < 16; ++e) {
            const int eg = ebase + e;
            const int s  = esrc[eg];
            const float Y1x = geo[wid][e][1], Y1y = geo[wid][e][2], Y1z = geo[wid][e][3];
            const float Y20 = geo[wid][e][4], Y21 = geo[wid][e][5], Y22 = geo[wid][e][6];
            const float Y23 = geo[wid][e][7], Y24 = geo[wid][e][8];
            const float w0 = wbuf[wid][e][c],        w1 = wbuf[wid][e][32 + c];
            const float w2 = wbuf[wid][e][64 + c],   w3 = wbuf[wid][e][96 + c];
            const float w4 = wbuf[wid][e][128 + c],  w5 = wbuf[wid][e][160 + c];
            const float w6 = wbuf[wid][e][192 + c];
            const float f0v = f0[(size_t)s * CC + c];
            const float* f1r = f1 + ((size_t)s * 3) * CC + c;
            const float* f2r = f2 + ((size_t)s * 5) * CC + c;
            const float f10 = f1r[0], f11 = f1r[CC], f12 = f1r[2*CC];
            const float f20 = f2r[0], f21 = f2r[CC], f22 = f2r[2*CC], f23 = f2r[3*CC], f24 = f2r[4*CC];
            const float dot1 = f10*Y1x + f11*Y1y + f12*Y1z;
            const float dot2 = f20*Y20 + f21*Y21 + f22*Y22 + f23*Y23 + f24*Y24;
            const float m0v = w0*f0v + w1*dot1 + w2*dot2;
            m0s[wid][e][c] = m0v;
            m0g[(size_t)eg * CC + c] = m0v;
            float* m1r = m1g + ((size_t)eg * 3) * CC + c;
            m1r[0]    = w3*f0v*Y1x + w4*f10;
            m1r[CC]   = w3*f0v*Y1y + w4*f11;
            m1r[2*CC] = w3*f0v*Y1z + w4*f12;
            float* m2r = m2g + ((size_t)eg * 5) * CC + c;
            m2r[0]    = w5*f0v*Y20 + w6*f20;
            m2r[CC]   = w5*f0v*Y21 + w6*f21;
            m2r[2*CC] = w5*f0v*Y22 + w6*f22;
            m2r[3*CC] = w5*f0v*Y23 + w6*f23;
            m2r[4*CC] = w5*f0v*Y24 + w6*f24;
        }
    }
    __syncthreads();

    // ---- q/k projections (WMMA) ----
    v16bf aM, aQ;
    #pragma unroll
    for (int j = 0; j < 8; ++j) { aM[j] = f2bf(m0s[wid][mrow][kh + j]); aM[8+j] = f2bf(m0s[wid][mrow][kh + 16 + j]); }
    {
        const int dn = edst[ebase + mrow];
        aQ = loadA_row(f0 + (size_t)dn * CC, lane);
    }
    {
        v8f z = {};
        v8f kd0 = wmma_bf16(aM, loadB(Wk, CC,  0, lane), z);
        v8f kd1 = wmma_bf16(aM, loadB(Wk, CC, 16, lane), z);
        v8f qd0 = wmma_bf16(aQ, loadB(Wq, CC,  0, lane), z);
        v8f qd1 = wmma_bf16(aQ, loadB(Wq, CC, 16, lane), z);
        const int nc = lane & 15;
        #pragma unroll
        for (int i = 0; i < 8; ++i) {
            const int m = i + hi * 8;
            qs[wid][m][nc] = qd0[i]; qs[wid][m][16 + nc] = qd1[i];
            ks[wid][m][nc] = kd0[i]; ks[wid][m][16 + nc] = kd1[i];
        }
    }
    __syncthreads();

    // ---- logits + segment-max (16 edges x 4 heads = 64 values) ----
    for (int idx = lane; idx < 64; idx += 32) {
        const int e = idx >> 2, h = idx & 3;
        float acc = 0.0f;
        #pragma unroll
        for (int j = 0; j < 8; ++j) acc += qs[wid][e][h*8 + j] * ks[wid][e][h*8 + j];
        acc *= 0.35355339059327373f;  // 1/sqrt(Dh=8)
        const int eg = ebase + e;
        logitsg[(size_t)eg * HH + h] = acc;
        atomicMaxF(&mx[(size_t)edst[eg] * HH + h], acc);
    }
}

// ---------------------------------------------------------------------------
__global__ __launch_bounds__(256) void exp_den_kernel(
    const int* __restrict__ edst, float* __restrict__ logitsg,
    const float* __restrict__ mx, float* __restrict__ den, int EHtot)
{
    const int t = blockIdx.x * 256 + threadIdx.x;
    if (t >= EHtot) return;
    const int e = t >> 2, h = t & 3;
    const int d = edst[e];
    float m = mx[(size_t)d * HH + h];
    if (!(m >= -3.0e38f && m <= 3.0e38f)) m = 0.0f;   // isfinite -> 0
    const float ev = expf(logitsg[t] - m);
    logitsg[t] = ev;                                   // overwrite with exp value
    atomicAdd(&den[(size_t)d * HH + h], ev);
}

__global__ __launch_bounds__(256) void aggregate_kernel(
    const int* __restrict__ edst, const float* __restrict__ logitsg,
    const float* __restrict__ den,
    const float* __restrict__ m0g, const float* __restrict__ m1g, const float* __restrict__ m2g,
    float* __restrict__ agg0, float* __restrict__ agg1, float* __restrict__ agg2, int ECtot)
{
    const int t = blockIdx.x * 256 + threadIdx.x;
    if (t >= ECtot) return;
    const int e = t >> 5, c = t & 31, h = c >> 3;
    const int d = edst[e];
    const float alpha = logitsg[(size_t)e * HH + h] / (den[(size_t)d * HH + h] + EPSF);
    atomicAdd(&agg0[(size_t)d * CC + c], alpha * m0g[(size_t)e * CC + c]);
    #pragma unroll
    for (int md = 0; md < 3; ++md)
        atomicAdd(&agg1[((size_t)d*3 + md) * CC + c], alpha * m1g[((size_t)e*3 + md) * CC + c]);
    #pragma unroll
    for (int md = 0; md < 5; ++md)
        atomicAdd(&agg2[((size_t)d*5 + md) * CC + c], alpha * m2g[((size_t)e*5 + md) * CC + c]);
}

// ---------------------------------------------------------------------------
// Node update: f0' = agg0@Wself0 + f0@Wskip ; f1' = (3N,32)@Wself1 ; f2' = (5N,32)@Wself2
// ---------------------------------------------------------------------------
__global__ __launch_bounds__(128) void node_update(
    const float* __restrict__ agg0, const float* __restrict__ agg1, const float* __restrict__ agg2,
    const float* __restrict__ f0old,
    const float* __restrict__ Wself0, const float* __restrict__ Wself1,
    const float* __restrict__ Wself2, const float* __restrict__ Wskip,
    float* __restrict__ f0new, float* __restrict__ f1new, float* __restrict__ f2new, int ntiles)
{
    const int wid = threadIdx.x >> 5, lane = threadIdx.x & 31;
    const int tile = blockIdx.x * 4 + wid;
    if (tile >= ntiles) return;
    const int row0 = tile * 16;
    v8f z = {};
    v16bf bs0[2], bsk[2], bs1[2], bs2[2];
    #pragma unroll
    for (int t = 0; t < 2; ++t) {
        bs0[t] = loadB(Wself0, CC, t * 16, lane);
        bsk[t] = loadB(Wskip,  CC, t * 16, lane);
        bs1[t] = loadB(Wself1, CC, t * 16, lane);
        bs2[t] = loadB(Wself2, CC, t * 16, lane);
    }
    // f0
    {
        v16bf aAgg = loadA_rows(agg0  + (size_t)row0 * CC, lane);
        v16bf aOld = loadA_rows(f0old + (size_t)row0 * CC, lane);
        #pragma unroll
        for (int t = 0; t < 2; ++t) {
            v8f d = wmma_bf16(aOld, bsk[t], z);
            d = wmma_bf16(aAgg, bs0[t], d);
            storeD(f0new + (size_t)row0 * CC, t, lane, d);
        }
    }
    // f1: 3 row-subtiles of 16
    #pragma unroll
    for (int sub = 0; sub < 3; ++sub) {
        const size_t rbase = ((size_t)tile * 48 + sub * 16) * CC;
        v16bf a = loadA_rows(agg1 + rbase, lane);
        #pragma unroll
        for (int t = 0; t < 2; ++t) storeD(f1new + rbase, t, lane, wmma_bf16(a, bs1[t], z));
    }
    // f2: 5 row-subtiles of 16
    #pragma unroll
    for (int sub = 0; sub < 5; ++sub) {
        const size_t rbase = ((size_t)tile * 80 + sub * 16) * CC;
        v16bf a = loadA_rows(agg2 + rbase, lane);
        #pragma unroll
        for (int t = 0; t < 2; ++t) storeD(f2new + rbase, t, lane, wmma_bf16(a, bs2[t], z));
    }
}

// ---------------------------------------------------------------------------
// Final: hs0 = f0 @ Wout ; cs = hs0 @ Wc (Wc is 32x15, pad N to 16)
// ---------------------------------------------------------------------------
__global__ __launch_bounds__(128) void final_out(
    const float* __restrict__ f0, const float* __restrict__ Wout, const float* __restrict__ Wc,
    float* __restrict__ out_hs, float* __restrict__ out_cs, int ntiles)
{
    __shared__ __bf16 hidb[4][16][32];
    const int wid = threadIdx.x >> 5, lane = threadIdx.x & 31;
    const int tile = blockIdx.x * 4 + wid;
    if (tile >= ntiles) return;
    const int row0 = tile * 16;
    const int hi = lane >> 4;
    v8f z = {};
    v16bf a = loadA_rows(f0 + (size_t)row0 * CC, lane);
    #pragma unroll
    for (int t = 0; t < 2; ++t) {
        v8f d = wmma_bf16(a, loadB(Wout, CC, t * 16, lane), z);
        const int col = t * 16 + (lane & 15);
        #pragma unroll
        for (int i = 0; i < 8; ++i) {
            const int m = i + hi * 8;
            out_hs[(size_t)(row0 + m) * CC + col] = d[i];
            hidb[wid][m][col] = f2bf(d[i]);
        }
    }
    // hs0 @ Wc
    v16bf a2;
    {
        const int mrow = lane & 15, kh = hi * 8;
        #pragma unroll
        for (int j = 0; j < 8; ++j) { a2[j] = hidb[wid][mrow][kh + j]; a2[8+j] = hidb[wid][mrow][kh + 16 + j]; }
    }
    v16bf bc;
    {
        const int n = lane & 15, kb = hi * 8;
        #pragma unroll
        for (int j = 0; j < 16; ++j) bc[j] = (n < 15) ? f2bf(Wc[(size_t)(kb + j + (j & 8)) * 15 + n]) : f2bf(0.0f);
    }
    v8f d = wmma_bf16(a2, bc, z);
    const int col = lane & 15;
    if (col < 15) {
        #pragma unroll
        for (int i = 0; i < 8; ++i) out_cs[(size_t)(row0 + i + hi * 8) * 15 + col] = d[i];
    }
}

__global__ __launch_bounds__(256) void fill_kernel(float* __restrict__ p, float v, size_t n) {
    const size_t t = (size_t)blockIdx.x * 256 + threadIdx.x;
    if (t < n) p[t] = v;
}

// ---------------------------------------------------------------------------
extern "C" void kernel_launch(void* const* d_in, const int* in_sizes, int n_in,
                              void* d_out, int out_size, void* d_ws, size_t ws_size,
                              hipStream_t stream) {
    (void)in_sizes; (void)n_in; (void)out_size; (void)ws_size;
    const float* pos       = (const float*)d_in[0];
    const float* node_l0   = (const float*)d_in[1];
    const float* edge_feat = (const float*)d_in[2];
    const int*   esrc      = (const int*)d_in[3];
    const int*   edst      = (const int*)d_in[4];
    const float* Wr1       = (const float*)d_in[5];
    const float* br1       = (const float*)d_in[6];
    const float* Wr2       = (const float*)d_in[7];
    const float* Wq        = (const float*)d_in[8];
    const float* Wk        = (const float*)d_in[9];
    const float* Wself0    = (const float*)d_in[10];
    const float* Wself1    = (const float*)d_in[11];
    const float* Wself2    = (const float*)d_in[12];
    const float* Wskip     = (const float*)d_in[13];
    const float* Wout      = (const float*)d_in[14];
    const float* Wc        = (const float*)d_in[15];

    float* w = (float*)d_ws;
    size_t o = 0;
    float* f0B  = w + o; o += (size_t)NN * 32;
    float* f1B  = w + o; o += (size_t)NN * 96;
    float* f2B  = w + o; o += (size_t)NN * 160;
    float* f0A  = w + o; o += (size_t)NN * 32;
    float* f1A  = w + o; o += (size_t)NN * 96;
    float* f2A  = w + o; o += (size_t)NN * 160;
    float* agg0 = w + o; o += (size_t)NN * 32;
    float* agg1 = w + o; o += (size_t)NN * 96;
    float* agg2 = w + o; o += (size_t)NN * 160;
    float* m0b  = w + o; o += (size_t)EE * 32;
    float* m1b  = w + o; o += (size_t)EE * 96;
    float* m2b  = w + o; o += (size_t)EE * 160;
    float* logb = w + o; o += (size_t)EE * 4;
    float* mx   = w + o; o += (size_t)NN * 4;
    float* den  = w + o; o += (size_t)NN * 4;

    const int ntiles = NN / 16;                 // 1875
    const int nblk   = (ntiles + 3) / 4;        // 4 waves/block
    const float NEG_INF = -__builtin_huge_valf();

    auto fill = [&](float* p, float v, size_t n) {
        fill_kernel<<<(unsigned)((n + 255) / 256), 256, 0, stream>>>(p, v, n);
    };

    // layer-0 f1/f2 inputs are zero (re-zero every launch; layer-1 writes them)
    fill(f1A, 0.0f, (size_t)NN * 96);
    fill(f2A, 0.0f, (size_t)NN * 160);

    for (int l = 0; l < LL; ++l) {
        const float* f0in = (l == 0) ? node_l0 : f0B;
        const float* f1in = (l == 0) ? f1A : f1B;
        const float* f2in = (l == 0) ? f2A : f2B;
        float* f0o = (l == 0) ? f0B : f0A;
        float* f1o = (l == 0) ? f1B : f1A;
        float* f2o = (l == 0) ? f2B : f2A;

        fill(mx,  NEG_INF, (size_t)NN * 4);
        fill(den, 0.0f,    (size_t)NN * 4);
        fill(agg0, 0.0f,   (size_t)NN * 32);
        fill(agg1, 0.0f,   (size_t)NN * 96);
        fill(agg2, 0.0f,   (size_t)NN * 160);

        edge_pass1<<<EE / (16 * WV), 32 * WV, 0, stream>>>(
            pos, f0in, f1in, f2in, edge_feat, esrc, edst,
            Wr1 + (size_t)l * 33 * 32, br1 + (size_t)l * 32,
            Wr2 + (size_t)l * 32 * 224, Wq + (size_t)l * 1024, Wk + (size_t)l * 1024,
            m0b, m1b, m2b, logb, mx);

        exp_den_kernel<<<(EE * 4 + 255) / 256, 256, 0, stream>>>(edst, logb, mx, den, EE * 4);

        aggregate_kernel<<<(EE * 32 + 255) / 256, 256, 0, stream>>>(
            edst, logb, den, m0b, m1b, m2b, agg0, agg1, agg2, EE * 32);

        node_update<<<nblk, 128, 0, stream>>>(
            agg0, agg1, agg2, f0in,
            Wself0 + (size_t)l * 1024, Wself1 + (size_t)l * 1024,
            Wself2 + (size_t)l * 1024, Wskip + (size_t)l * 1024,
            f0o, f1o, f2o, ntiles);
    }

    float* out_hs = (float*)d_out;
    float* out_cs = out_hs + (size_t)NN * 32;
    final_out<<<nblk, 128, 0, stream>>>(f0A, Wout, Wc, out_hs, out_cs, ntiles);
}
